// ProteinGATModel_35107062678098
// MI455X (gfx1250) — matrix operand order repaired and checked by reading, not compiled, not run on hardware
//
#include <hip/hip_runtime.h>
#include <hip/hip_bf16.h>

#define N_NODES 50000
#define N_EDGES 800000
#define NBATCH  128
#define NFEAT   7
#define EDIM    2
#define HID     128
#define HEADS   4
#define OUT_DIM 64
#define NLAYERS 3
#define E2      (N_EDGES + N_NODES)
#define EPSV    1e-5f
#define NEG_SLOPE 0.2f
#define CDIV(a,b) (((a)+(b)-1)/(b))

typedef __attribute__((ext_vector_type(16))) __bf16 v16bf;
typedef __attribute__((ext_vector_type(8)))  float  v8f;

union Frag { v16bf v; uint4 u[2]; };

// ---- order-preserving float <-> uint key (for atomic segment max) ----
__device__ __forceinline__ unsigned f2key(float f) {
  unsigned u = __float_as_uint(f);
  return (u & 0x80000000u) ? ~u : (u | 0x80000000u);
}
__device__ __forceinline__ float key2f(unsigned k) {
  unsigned u = (k & 0x80000000u) ? (k ^ 0x80000000u) : ~k;
  return __uint_as_float(u);
}

// =====================================================================
// Weight pre-pack: W[f32, 128 x ncols row-major] -> Wp[bf16, fragment order]
// Fragment order (B 32x16 per wave32, contiguous k within each lane half; A uses the same):
//   Wp[((ntile*4 + kk)*32 + lane)*16 + e]
//   where k = kk*32 + (lane>=16)*16 + e ; col = ntile*16 + (lane&15)
// Each lane's 16 elements are 32 contiguous bytes -> 2x b128 loads in GEMM.
// =====================================================================
__global__ void k_pack_w(const float* __restrict__ W, __bf16* __restrict__ Wp,
                         int ncols) {
  const int idx = blockIdx.x * blockDim.x + threadIdx.x;
  if (idx >= HID * ncols) return;
  const int e    = idx & 15;
  const int lane = (idx >> 4) & 31;
  const int kk   = (idx >> 9) & 3;
  const int nt   = idx >> 11;
  const int k    = kk * 32 + (lane >> 4) * 16 + e;
  const int col  = nt * 16 + (lane & 15);
  Wp[idx] = (__bf16)W[k * ncols + col];
}

// =====================================================================
// WMMA bf16 GEMM: C[M x ncols] = A[M x 128] * W[128 x ncols]
// block = 128 threads (4 waves); block computes 64 rows x up to 128 cols.
// A staged in LDS in fragment order (vectorized both directions);
// B read from pre-packed global (2x b128 per fragment).
// 4 M-tiles per wave -> 32 v_wmma per wave, B fragments reused 4x.
// =====================================================================
__global__ __launch_bounds__(128)
void k_gemm_bf16(const __bf16* __restrict__ A, const __bf16* __restrict__ Wp,
                 float* __restrict__ C, int M, int ncols) {
  // Ap: 4 mtiles x 4 kk x 32 lanes x 16 elems (bf16) = 16 KB, fragment order
  __shared__ __align__(16) __bf16 Ap[4 * 4 * 32 * 16];
  const int row0 = blockIdx.x * 64;
  const int tid  = threadIdx.x;

  // cooperative load: 1024 chunks of 16B; chunk c:
  //   mt=c>>8, kk=(c>>6)&3, lane=(c>>1)&31, half=c&1
  //   src row = mt*16 + (lane&15); src k = kk*32 + (lane>>4)*16 + half*8
#pragma unroll
  for (int t = 0; t < 8; ++t) {
    const int c    = t * 128 + tid;
    const int mt   = c >> 8;
    const int kk   = (c >> 6) & 3;
    const int lane = (c >> 1) & 31;
    const int half = c & 1;
    const int row  = mt * 16 + (lane & 15);
    const int ks   = kk * 32 + (lane >> 4) * 16 + half * 8;
    uint4 val = {0u, 0u, 0u, 0u};
    if (row0 + row < M)
      val = *(const uint4*)(A + (size_t)(row0 + row) * HID + ks);
    *(uint4*)(Ap + (size_t)c * 8) = val;
  }
  // prefetch next block's A tile (global_prefetch_b8 path)
  if (row0 + 64 + (tid >> 1) < M)
    __builtin_prefetch(A + (size_t)(row0 + 64 + (tid >> 1)) * HID + (tid & 1) * 64, 0, 1);
  __syncthreads();

  const int wave = tid >> 5;
  const int lane = tid & 31;
  const int col0 = wave * 32;
  if (col0 >= ncols) return;
  const int laneM = lane & 15;
  const int hi    = lane >> 4;

  v8f acc0[4], acc1[4];
#pragma unroll
  for (int mt = 0; mt < 4; ++mt) { acc0[mt] = (v8f){}; acc1[mt] = (v8f){}; }

#pragma unroll
  for (int kk = 0; kk < 4; ++kk) {
    Frag b0, b1;
    {
      const uint4* p0 = (const uint4*)(Wp + (((size_t)(wave * 2) * 4 + kk) * 32 + lane) * 16);
      const uint4* p1 = (const uint4*)(Wp + (((size_t)(wave * 2 + 1) * 4 + kk) * 32 + lane) * 16);
      b0.u[0] = p0[0]; b0.u[1] = p0[1];
      b1.u[0] = p1[0]; b1.u[1] = p1[1];
    }
#pragma unroll
    for (int mt = 0; mt < 4; ++mt) {
      Frag a;
      const uint4* ap = (const uint4*)(Ap + ((size_t)(mt * 4 + kk) * 32 + lane) * 16);
      a.u[0] = ap[0]; a.u[1] = ap[1];
      acc0[mt] = __builtin_amdgcn_wmma_f32_16x16x32_bf16(false, a.v, false, b0.v,
                                                         (short)0, acc0[mt], false, false);
      acc1[mt] = __builtin_amdgcn_wmma_f32_16x16x32_bf16(false, a.v, false, b1.v,
                                                         (short)0, acc1[mt], false, false);
    }
  }

  // C 16x16: lanes 0-15 rows M=0..7, lanes 16-31 rows M=8..15; N=lane&15
#pragma unroll
  for (int mt = 0; mt < 4; ++mt) {
#pragma unroll
    for (int r = 0; r < 8; ++r) {
      const int mr = mt * 16 + hi * 8 + r;
      if (row0 + mr < M) {
        C[(size_t)(row0 + mr) * ncols + col0 + laneM]      = acc0[mt][r];
        C[(size_t)(row0 + mr) * ncols + col0 + laneM + 16] = acc1[mt][r];
      }
    }
  }
}

// =====================================================================
// Node encoder: h = LayerNorm(ReLU(x @ encW + encb)), one block per node
// =====================================================================
__global__ __launch_bounds__(128)
void k_encoder(const float* __restrict__ x, const float* __restrict__ W,
               const float* __restrict__ b, const float* __restrict__ g,
               const float* __restrict__ bb, float* __restrict__ h) {
  const int n = blockIdx.x, d = threadIdx.x;
  float acc = b[d];
#pragma unroll
  for (int k = 0; k < NFEAT; ++k) acc += x[n * NFEAT + k] * W[k * HID + d];
  float val = fmaxf(acc, 0.f);
  __shared__ float sh[HID];
  sh[d] = val; __syncthreads();
  for (int s = 64; s > 0; s >>= 1) { if (d < s) sh[d] += sh[d + s]; __syncthreads(); }
  const float mu = sh[0] * (1.f / HID);
  __syncthreads();
  const float xc = val - mu;
  sh[d] = xc * xc; __syncthreads();
  for (int s = 64; s > 0; s >>= 1) { if (d < s) sh[d] += sh[d + s]; __syncthreads(); }
  const float var = sh[0] * (1.f / HID);
  h[(size_t)n * HID + d] = g[d] * xc * rsqrtf(var + EPSV) + bb[d];
}

// ReLU+LayerNorm row kernel (mapper): out = LN(relu(in+bias))
__global__ __launch_bounds__(128)
void k_relu_ln_row(const float* __restrict__ in, const float* __restrict__ bias,
                   const float* __restrict__ g, const float* __restrict__ bb,
                   float* __restrict__ out) {
  const int n = blockIdx.x, d = threadIdx.x;
  float val = fmaxf(in[n * HID + d] + bias[d], 0.f);
  __shared__ float sh[HID];
  sh[d] = val; __syncthreads();
  for (int s = 64; s > 0; s >>= 1) { if (d < s) sh[d] += sh[d + s]; __syncthreads(); }
  const float mu = sh[0] * (1.f / HID);
  __syncthreads();
  const float xc = val - mu;
  sh[d] = xc * xc; __syncthreads();
  for (int s = 64; s > 0; s >>= 1) { if (d < s) sh[d] += sh[d + s]; __syncthreads(); }
  out[n * HID + d] = g[d] * xc * rsqrtf(sh[0] * (1.f / HID) + EPSV) + bb[d];
}

// =====================================================================
// Edge pass 1: GATv2 scores + segment max (one wave per edge, wave32)
// =====================================================================
__global__ __launch_bounds__(256)
void k_edge_score(const float* __restrict__ xs, const float* __restrict__ xd,
                  const int* __restrict__ ei, const float* __restrict__ eattr,
                  const float* __restrict__ easum,
                  const float* __restrict__ We, const float* __restrict__ att,
                  float* __restrict__ score, unsigned* __restrict__ mmax) {
  const int e = (blockIdx.x * blockDim.x + threadIdx.x) >> 5;
  const int lane = threadIdx.x & 31;
  if (e >= E2) return;
  int src, dst; float ea0, ea1;
  if (e < N_EDGES) {
    src = ei[e]; dst = ei[N_EDGES + e];
    ea0 = eattr[2 * e]; ea1 = eattr[2 * e + 1];
  } else {
    src = dst = e - N_EDGES;
    ea0 = easum[0] * (1.f / N_EDGES); ea1 = easum[1] * (1.f / N_EDGES);
  }
  float p = 0.f;
#pragma unroll
  for (int j = 0; j < 4; ++j) {
    const int d = lane * 4 + j;
    float z = xs[(size_t)src * HID + d] + xd[(size_t)dst * HID + d]
            + ea0 * We[d] + ea1 * We[HID + d];
    z = (z > 0.f) ? z : NEG_SLOPE * z;
    p += z * att[d];
  }
  p += __shfl_xor(p, 1);
  p += __shfl_xor(p, 2);
  p += __shfl_xor(p, 4);
  if ((lane & 7) == 0) {
    const int hh = lane >> 3;
    score[(size_t)e * HEADS + hh] = p;
    atomicMax(&mmax[(size_t)dst * HEADS + hh], f2key(p));
  }
}

// Edge pass 2: ex = exp(score - max[dst]); denom[dst] += ex
__global__ __launch_bounds__(256)
void k_edge_exp(float* __restrict__ score, const unsigned* __restrict__ mmax,
                float* __restrict__ denom, const int* __restrict__ ei) {
  const int idx = blockIdx.x * blockDim.x + threadIdx.x;
  if (idx >= E2 * HEADS) return;
  const int e = idx >> 2, hh = idx & 3;
  const int dst = (e < N_EDGES) ? ei[N_EDGES + e] : (e - N_EDGES);
  const float ex = __expf(score[idx] - key2f(mmax[dst * HEADS + hh]));
  score[idx] = ex;
  atomicAdd(&denom[dst * HEADS + hh], ex);
}

// Edge pass 3: agg[dst] += xs[src] * alpha  (one wave per edge)
__global__ __launch_bounds__(256)
void k_edge_agg(const float* __restrict__ xs, const int* __restrict__ ei,
                const float* __restrict__ score, const float* __restrict__ denom,
                float* __restrict__ agg) {
  const int e = (blockIdx.x * blockDim.x + threadIdx.x) >> 5;
  const int lane = threadIdx.x & 31;
  if (e >= E2) return;
  const int src = (e < N_EDGES) ? ei[e] : (e - N_EDGES);
  const int dst = (e < N_EDGES) ? ei[N_EDGES + e] : (e - N_EDGES);
  const int hh = lane >> 3;
  const float a = score[(size_t)e * HEADS + hh] / denom[(size_t)dst * HEADS + hh];
#pragma unroll
  for (int j = 0; j < 4; ++j) {
    const int d = lane * 4 + j;
    atomicAdd(&agg[(size_t)dst * HID + d], xs[(size_t)src * HID + d] * a);
  }
}

// =====================================================================
// GraphNorm (3 streaming passes + atomics into per-batch stats)
// =====================================================================
__global__ __launch_bounds__(256)
void k_gn_a(float* __restrict__ agg, const float* __restrict__ bias,
            const float* __restrict__ h, const int* __restrict__ batch,
            float* __restrict__ gsum) {
  const int idx = blockIdx.x * blockDim.x + threadIdx.x;
  if (idx >= N_NODES * HID) return;
  const int n = idx >> 7, d = idx & 127;
  const float t = agg[idx] + bias[d] + h[idx];   // + residual
  agg[idx] = t;
  atomicAdd(&gsum[batch[n] * HID + d], t);
}
__global__ __launch_bounds__(256)
void k_gn_b(float* __restrict__ agg, const float* __restrict__ gsum,
            const float* __restrict__ cnt, const float* __restrict__ sscale,
            const int* __restrict__ batch, float* __restrict__ gvar) {
  const int idx = blockIdx.x * blockDim.x + threadIdx.x;
  if (idx >= N_NODES * HID) return;
  const int n = idx >> 7, d = idx & 127;
  const int b = batch[n];
  const float c = fmaxf(cnt[b], 1.f);
  const float xc = agg[idx] - sscale[d] * (gsum[b * HID + d] / c);
  agg[idx] = xc;
  atomicAdd(&gvar[b * HID + d], xc * xc);
}
__global__ __launch_bounds__(256)
void k_gn_c(const float* __restrict__ agg, const float* __restrict__ gvar,
            const float* __restrict__ cnt, const float* __restrict__ wgt,
            const float* __restrict__ bb, const int* __restrict__ batch,
            float* __restrict__ h) {
  const int idx = blockIdx.x * blockDim.x + threadIdx.x;
  if (idx >= N_NODES * HID) return;
  const int n = idx >> 7, d = idx & 127;
  const int b = batch[n];
  const float c = fmaxf(cnt[b], 1.f);
  const float y = wgt[d] * agg[idx] * rsqrtf(gvar[b * HID + d] / c + EPSV) + bb[d];
  h[idx] = (y > 0.f) ? y : (__expf(y) - 1.f);    // ELU
}

// =====================================================================
// Attention pooling
// =====================================================================
__global__ __launch_bounds__(128)
void k_pool_score(const float* __restrict__ ph, const float* __restrict__ b1,
                  const float* __restrict__ W2, float* __restrict__ s,
                  unsigned* __restrict__ smax) {
  const int n = blockIdx.x, d = threadIdx.x;
  const float t = tanhf(ph[(size_t)n * HID + d] + b1[d]) * W2[d];
  __shared__ float sh[HID];
  sh[d] = t; __syncthreads();
  for (int st = 64; st > 0; st >>= 1) { if (d < st) sh[d] += sh[d + st]; __syncthreads(); }
  if (d == 0) { s[n] = sh[0]; atomicMax(smax, f2key(sh[0])); }
}
__global__ __launch_bounds__(256)
void k_pool_exp(float* __restrict__ s, const unsigned* __restrict__ smax,
                float* __restrict__ ssum) {
  const int i = blockIdx.x * 256 + threadIdx.x;
  float w = 0.f;
  if (i < N_NODES) { w = __expf(s[i] - key2f(*smax)); s[i] = w; }
  __shared__ float sh[256];
  sh[threadIdx.x] = w; __syncthreads();
  for (int st = 128; st > 0; st >>= 1) { if (threadIdx.x < st) sh[threadIdx.x] += sh[threadIdx.x + st]; __syncthreads(); }
  if (threadIdx.x == 0) atomicAdd(ssum, sh[0]);
}
__global__ __launch_bounds__(256)
void k_pool_agg(const float* __restrict__ h, const float* __restrict__ s,
                const float* __restrict__ ssum, const int* __restrict__ batch,
                float* __restrict__ g) {
  const int idx = blockIdx.x * blockDim.x + threadIdx.x;
  if (idx >= N_NODES * HID) return;
  const int n = idx >> 7, d = idx & 127;
  atomicAdd(&g[batch[n] * HID + d], h[idx] * s[n] * (1.f / *ssum));
}

// =====================================================================
// utility kernels
// =====================================================================
__global__ void k_zero_f32(float* p, int n) {
  int i = blockIdx.x * blockDim.x + threadIdx.x; if (i < n) p[i] = 0.f;
}
__global__ void k_fill_u32(unsigned* p, unsigned v, int n) {
  int i = blockIdx.x * blockDim.x + threadIdx.x; if (i < n) p[i] = v;
}
// vectorized f32 -> bf16 (4 at a time; all our sizes are multiples of 4)
__global__ void k_cvt_bf16(const float* __restrict__ in, __bf16* __restrict__ out, int n) {
  const int i = (blockIdx.x * blockDim.x + threadIdx.x) * 4;
  if (i + 3 < n) {
    const float4 f = *(const float4*)(in + i);
    union { __bf16 h[4]; uint2 u; } o;
    o.h[0] = (__bf16)f.x; o.h[1] = (__bf16)f.y;
    o.h[2] = (__bf16)f.z; o.h[3] = (__bf16)f.w;
    *(uint2*)(out + i) = o.u;
  } else {
    for (int j = 0; j < 4 && i + j < n; ++j) out[i + j] = (__bf16)in[i + j];
  }
}
__global__ void k_copy_f32(float* __restrict__ dst, const float* __restrict__ src, int n) {
  int i = blockIdx.x * blockDim.x + threadIdx.x; if (i < n) dst[i] = src[i];
}
__global__ void k_count(const int* __restrict__ batch, float* __restrict__ cnt) {
  int i = blockIdx.x * blockDim.x + threadIdx.x;
  if (i < N_NODES) atomicAdd(&cnt[batch[i]], 1.f);
}
__global__ __launch_bounds__(256)
void k_ea_sum(const float* __restrict__ ea, float* __restrict__ easum) {
  int i = blockIdx.x * 256 + threadIdx.x;
  float a0 = 0.f, a1 = 0.f;
  if (i < N_EDGES) { a0 = ea[2 * i]; a1 = ea[2 * i + 1]; }
  for (int m = 1; m < 32; m <<= 1) { a0 += __shfl_xor(a0, m); a1 += __shfl_xor(a1, m); }
  if ((threadIdx.x & 31) == 0) { atomicAdd(&easum[0], a0); atomicAdd(&easum[1], a1); }
}
__global__ void k_out_bias(const float* __restrict__ t, const float* __restrict__ b,
                           float* __restrict__ out) {
  int i = blockIdx.x * blockDim.x + threadIdx.x;
  if (i < NBATCH * OUT_DIM) out[i] = t[i] + b[i % OUT_DIM];
}

// =====================================================================
extern "C" void kernel_launch(void* const* d_in, const int* in_sizes, int n_in,
                              void* d_out, int out_size, void* d_ws, size_t ws_size,
                              hipStream_t stream) {
  (void)in_sizes; (void)n_in; (void)out_size; (void)ws_size;
  const float* x      = (const float*)d_in[0];
  const int*   ei     = (const int*)d_in[1];
  const float* eattr  = (const float*)d_in[2];
  const int*   batch  = (const int*)d_in[3];
  const float* encW   = (const float*)d_in[4];
  const float* encb   = (const float*)d_in[5];
  const float* encg   = (const float*)d_in[6];
  const float* encbb  = (const float*)d_in[7];
  const float* gWsrc  = (const float*)d_in[8];
  const float* gWdst  = (const float*)d_in[9];
  const float* gWe    = (const float*)d_in[10];
  const float* gatt   = (const float*)d_in[11];
  const float* gbias  = (const float*)d_in[12];
  const float* gnw    = (const float*)d_in[13];
  const float* gnb    = (const float*)d_in[14];
  const float* gns    = (const float*)d_in[15];
  const float* poolW1 = (const float*)d_in[16];
  const float* poolb1 = (const float*)d_in[17];
  const float* poolW2 = (const float*)d_in[18];
  const float* mapW1  = (const float*)d_in[19];
  const float* mapb1  = (const float*)d_in[20];
  const float* mapg   = (const float*)d_in[21];
  const float* mapbb  = (const float*)d_in[22];
  const float* mapW2  = (const float*)d_in[23];
  const float* mapb2  = (const float*)d_in[24];
  float* out = (float*)d_out;

  // ---- workspace carve ----
  char* wp = (char*)d_ws;
  auto carve = [&](size_t bytes) -> char* {
    char* p = wp; wp += (bytes + 255) & ~(size_t)255; return p;
  };
  float*    h     = (float*)carve((size_t)N_NODES * HID * 4);
  __bf16*   hbf   = (__bf16*)carve((size_t)N_NODES * HID * 2);
  float*    xs    = (float*)carve((size_t)N_NODES * HID * 4);   // also pool ph
  float*    xd    = (float*)carve((size_t)N_NODES * HID * 4);
  float*    agg   = (float*)carve((size_t)N_NODES * HID * 4);
  float*    score = (float*)carve((size_t)E2 * HEADS * 4);
  unsigned* mmax  = (unsigned*)carve((size_t)N_NODES * HEADS * 4);
  float*    denom = (float*)carve((size_t)N_NODES * HEADS * 4);
  float*    cnt   = (float*)carve(NBATCH * 4);
  float*    gsum  = (float*)carve(NBATCH * HID * 4);
  float*    gvar  = (float*)carve(NBATCH * HID * 4);
  float*    easum = (float*)carve(2 * 4);
  float*    psc   = (float*)carve(N_NODES * 4);
  unsigned* smax  = (unsigned*)carve(4);
  float*    ssum  = (float*)carve(4);
  float*    gpool = (float*)carve(NBATCH * HID * 4);
  __bf16*   gbf   = (__bf16*)carve(NBATCH * HID * 2);
  float*    t1    = (float*)carve(NBATCH * HID * 4);
  float*    t2    = (float*)carve(NBATCH * HID * 4);
  __bf16*   t2bf  = (__bf16*)carve(NBATCH * HID * 2);
  float*    t3    = (float*)carve(NBATCH * OUT_DIM * 4);
  __bf16*   wbfA  = (__bf16*)carve(HID * HID * 2);   // packed weights
  __bf16*   wbfB  = (__bf16*)carve(HID * HID * 2);

  const int NH = N_NODES * HID;

  // ---- one-time stats ----
  k_zero_f32<<<CDIV(NBATCH, 256), 256, 0, stream>>>(cnt, NBATCH);
  k_zero_f32<<<1, 2, 0, stream>>>(easum, 2);
  k_count<<<CDIV(N_NODES, 256), 256, 0, stream>>>(batch, cnt);
  k_ea_sum<<<CDIV(N_EDGES, 256), 256, 0, stream>>>(eattr, easum);

  // ---- encoder ----
  k_encoder<<<N_NODES, 128, 0, stream>>>(x, encW, encb, encg, encbb, h);

  // ---- GATv2 layers ----
  for (int i = 0; i < NLAYERS; ++i) {
    const float* Wsrc = gWsrc + (size_t)i * HID * HID;
    const float* Wdst = gWdst + (size_t)i * HID * HID;
    const float* We   = gWe + (size_t)i * EDIM * HID;
    const float* att  = gatt + (size_t)i * HEADS * 32;
    const float* bias = gbias + (size_t)i * HID;

    k_cvt_bf16<<<CDIV(NH, 1024), 256, 0, stream>>>(h, hbf, NH);
    k_pack_w<<<CDIV(HID * HID, 256), 256, 0, stream>>>(Wsrc, wbfA, HID);
    k_pack_w<<<CDIV(HID * HID, 256), 256, 0, stream>>>(Wdst, wbfB, HID);
    k_gemm_bf16<<<CDIV(N_NODES, 64), 128, 0, stream>>>(hbf, wbfA, xs, N_NODES, HID);
    k_gemm_bf16<<<CDIV(N_NODES, 64), 128, 0, stream>>>(hbf, wbfB, xd, N_NODES, HID);

    k_fill_u32<<<CDIV(N_NODES * HEADS, 256), 256, 0, stream>>>(mmax, 0u, N_NODES * HEADS);
    k_zero_f32<<<CDIV(N_NODES * HEADS, 256), 256, 0, stream>>>(denom, N_NODES * HEADS);
    k_zero_f32<<<CDIV(NH, 256), 256, 0, stream>>>(agg, NH);

    k_edge_score<<<CDIV(E2 * 32, 256), 256, 0, stream>>>(xs, xd, ei, eattr, easum, We, att, score, mmax);
    k_edge_exp<<<CDIV(E2 * HEADS, 256), 256, 0, stream>>>(score, mmax, denom, ei);
    k_edge_agg<<<CDIV(E2 * 32, 256), 256, 0, stream>>>(xs, ei, score, denom, agg);

    k_zero_f32<<<CDIV(NBATCH * HID, 256), 256, 0, stream>>>(gsum, NBATCH * HID);
    k_zero_f32<<<CDIV(NBATCH * HID, 256), 256, 0, stream>>>(gvar, NBATCH * HID);
    k_gn_a<<<CDIV(NH, 256), 256, 0, stream>>>(agg, bias, h, batch, gsum);
    k_gn_b<<<CDIV(NH, 256), 256, 0, stream>>>(agg, gsum, cnt, gns + (size_t)i * HID, batch, gvar);
    k_gn_c<<<CDIV(NH, 256), 256, 0, stream>>>(agg, gvar, cnt, gnw + (size_t)i * HID,
                                              gnb + (size_t)i * HID, batch, h);
  }

  // ---- attention pooling ----
  k_cvt_bf16<<<CDIV(NH, 1024), 256, 0, stream>>>(h, hbf, NH);
  k_pack_w<<<CDIV(HID * HID, 256), 256, 0, stream>>>(poolW1, wbfA, HID);
  k_gemm_bf16<<<CDIV(N_NODES, 64), 128, 0, stream>>>(hbf, wbfA, xs, N_NODES, HID); // ph in xs
  k_fill_u32<<<1, 1, 0, stream>>>(smax, 0u, 1);
  k_zero_f32<<<1, 1, 0, stream>>>(ssum, 1);
  k_zero_f32<<<CDIV(NBATCH * HID, 256), 256, 0, stream>>>(gpool, NBATCH * HID);
  k_pool_score<<<N_NODES, 128, 0, stream>>>(xs, poolb1, poolW2, psc, smax);
  k_pool_exp<<<CDIV(N_NODES, 256), 256, 0, stream>>>(psc, smax, ssum);
  k_pool_agg<<<CDIV(NH, 256), 256, 0, stream>>>(h, psc, ssum, batch, gpool);

  // ---- output mapper ----
  k_cvt_bf16<<<CDIV(NBATCH * HID, 1024), 256, 0, stream>>>(gpool, gbf, NBATCH * HID);
  k_pack_w<<<CDIV(HID * HID, 256), 256, 0, stream>>>(mapW1, wbfA, HID);
  k_gemm_bf16<<<CDIV(NBATCH, 64), 128, 0, stream>>>(gbf, wbfA, t1, NBATCH, HID);
  k_relu_ln_row<<<NBATCH, 128, 0, stream>>>(t1, mapb1, mapg, mapbb, t2);
  k_cvt_bf16<<<CDIV(NBATCH * HID, 1024), 256, 0, stream>>>(t2, t2bf, NBATCH * HID);
  k_pack_w<<<CDIV(HID * OUT_DIM, 256), 256, 0, stream>>>(mapW2, wbfB, OUT_DIM);
  k_gemm_bf16<<<CDIV(NBATCH, 64), 128, 0, stream>>>(t2bf, wbfB, t3, NBATCH, OUT_DIM);
  k_out_bias<<<CDIV(NBATCH * OUT_DIM, 256), 256, 0, stream>>>(t3, mapb2, out);

  // ---- node embeddings (second tuple output) ----
  k_copy_f32<<<CDIV(NH, 256), 256, 0, stream>>>(out + NBATCH * OUT_DIM, h, NH);
}
